// MultiHeadAttention_81501299409410
// MI455X (gfx1250) — matrix operand order repaired and checked by reading, not compiled, not run on hardware
//
#include <hip/hip_runtime.h>

typedef __attribute__((ext_vector_type(16))) __bf16 v16bf;
typedef __attribute__((ext_vector_type(8)))  __bf16 v8bf;
typedef __attribute__((ext_vector_type(8)))  float  v8f;

namespace {
constexpr int kC    = 512;           // N_EMBED
constexpr int kH    = 16;            // heads
constexpr int kD    = 32;            // head dim (== WMMA bf16 K depth)
constexpr int kT    = 2048;          // sequence length
constexpr int kB    = 4;             // batch
constexpr int kRows = kB * kT;       // 8192
constexpr int kLD   = 48;            // padded LDS row stride (elements, 96B = 16B-aligned)
}

// (1/sqrt(32)) * log2(e) : fold softmax scale + exp->exp2 conversion into q
#define QSCALE 0.2550181753284786f

union abfrag { v16bf v; v8bf h[2]; };

// A-matrix 16x32 bf16 frag: lane m=lane%16, khalf=lane/16
// holds K = {kh*8 .. +7} and {16+kh*8 .. +7} (ISA 7.12.2)
__device__ __forceinline__ v16bf load_a_frag(const __bf16* row, int kh) {
  abfrag f;
  f.h[0] = *(const v8bf*)(row + kh * 8);
  f.h[1] = *(const v8bf*)(row + 16 + kh * 8);
  return f.v;
}
// B-matrix 32x16 bf16 frag: lane n=lane%16 holds its column's K = {kh*8 .. +7} and {16+kh*8 .. +7}
__device__ __forceinline__ v16bf load_b_frag(const __bf16* col, int kh) {
  abfrag f;
  f.h[0] = *(const v8bf*)(col + kh * 8);
  f.h[1] = *(const v8bf*)(col + 16 + kh * 8);
  return f.v;
}

__device__ __forceinline__ v8f vzero8() {
  v8f z = {0.f, 0.f, 0.f, 0.f, 0.f, 0.f, 0.f, 0.f};
  return z;
}

__device__ __forceinline__ v8f wmma_bf16(v16bf a, v16bf b, v8f c) {
  // (neg_a, A, neg_b, B, c_mod, C, reuse_a, reuse_b)
  return __builtin_amdgcn_wmma_f32_16x16x32_bf16(false, a, false, b, (short)0, c,
                                                 false, false);
}

// ---------------------------------------------------------------------------
// CDNA5 async global->LDS copy (ISA 10.x / 15.18, GVS mode), ASYNCcnt-tracked.
// vdst = wave-relative LDS byte address; vaddr = signed 32-bit byte offset;
// saddr = 64-bit uniform base. Bypasses VGPRs entirely.
// ---------------------------------------------------------------------------
__device__ __forceinline__ void async_load_b128(unsigned lds_byte_off,
                                                unsigned global_byte_off,
                                                const void* base) {
  asm volatile("global_load_async_to_lds_b128 %0, %1, %2"
               :
               : "v"(lds_byte_off), "v"(global_byte_off),
                 "s"((unsigned long long)(size_t)base)
               : "memory");
}
__device__ __forceinline__ void wait_asynccnt0() {
  asm volatile("s_wait_asynccnt 0x0" ::: "memory");
}
__device__ __forceinline__ unsigned lds_off_of(const void* p) {
  // generic LDS pointer keeps the wave-relative LDS offset in addr[31:0]
  return (unsigned)(size_t)p;
}

__global__ void __launch_bounds__(256) cast_f32_to_bf16(const float* __restrict__ in,
                                                        __bf16* __restrict__ out,
                                                        int n) {
  int i = blockIdx.x * 256 + threadIdx.x;
  if (i < n) out[i] = (__bf16)in[i];
}

// ---------------------------------------------------------------------------
// Shared 128x128 GEMM core: C[128x128] += A[128xK] * B[Kx128], bf16 in, f32 acc.
// 8 waves in 4(m) x 2(n) grid; each wave owns 32x64 = 2x4 WMMA tiles.
// A tile staged via async-to-LDS copies; B tile transposed via DS scatter.
// ---------------------------------------------------------------------------
__device__ __forceinline__ void gemm_core_128x128(
    const __bf16* __restrict__ A, const __bf16* __restrict__ B, int N, int K,
    int bm, int bn, __bf16* ldsA, __bf16* ldsBT, v8f acc[2][4]) {
  const int tid  = threadIdx.x;
  const int lane = tid & 31;
  const int wave = tid >> 5;
  const int wm   = wave >> 1;  // 0..3
  const int wn   = wave & 1;   // 0..1
  const int kh   = lane >> 4;
  const int lm   = lane & 15;

#pragma unroll
  for (int im = 0; im < 2; ++im)
#pragma unroll
    for (int in = 0; in < 4; ++in) acc[im][in] = vzero8();

  // per-thread staging coordinates (loop-invariant)
  const int arow  = tid >> 1;           // A: 128 rows, 2 threads/row
  const int akoff = (tid & 1) * 16;     // elements
  const unsigned alds = lds_off_of(ldsA + arow * kLD + akoff);
  const int bk  = tid >> 3;             // B: 32 k-rows, 8 threads/row
  const int bn0 = (tid & 7) * 16;       // elements

  for (int k0 = 0; k0 < K; k0 += 32) {
    {  // A tile [128 x 32] -> ldsA via async DMA (2 x 16B per thread)
      const unsigned goff =
          (unsigned)(((size_t)(bm + arow) * K + k0 + akoff) * sizeof(__bf16));
      async_load_b128(alds, goff, A);
      async_load_b128(alds + 16, goff + 16, A);
    }
    {  // B tile rows k0..+31, cols bn..+127 -> transposed ldsBT[n][k]
      const __bf16* g = B + (size_t)(k0 + bk) * N + bn + bn0;
#pragma unroll
      for (int j = 0; j < 16; ++j) ldsBT[(bn0 + j) * kLD + bk] = g[j];
    }
    if (k0 + 32 < K) {  // pull next K-step tiles toward caches
      __builtin_prefetch(A + (size_t)(bm + arow) * K + (k0 + 32) + akoff, 0, 0);
      __builtin_prefetch(B + (size_t)(k0 + 32 + bk) * N + bn + bn0, 0, 0);
    }
    wait_asynccnt0();
    __syncthreads();

    v16bf af[2], bfr[4];
#pragma unroll
    for (int im = 0; im < 2; ++im)
      af[im] = load_a_frag(ldsA + (wm * 32 + im * 16 + lm) * kLD, kh);
#pragma unroll
    for (int in = 0; in < 4; ++in)
      bfr[in] = load_b_frag(ldsBT + (wn * 64 + in * 16 + lm) * kLD, kh);
#pragma unroll
    for (int im = 0; im < 2; ++im)
#pragma unroll
      for (int in = 0; in < 4; ++in)
        acc[im][in] = wmma_bf16(af[im], bfr[in], acc[im][in]);
    __syncthreads();
  }
}

// ---------------------------------------------------------------------------
// QKV projection: [8192,512] x [512,1536] + bias, scattered into per-head
// bf16 q/k/v [B,H,T,32]; q pre-scaled by (1/sqrt(D))*log2(e).
// ---------------------------------------------------------------------------
__global__ void __launch_bounds__(256) qkv_gemm_kernel(
    const __bf16* __restrict__ xb, const __bf16* __restrict__ wab,
    const float* __restrict__ b_attn, __bf16* __restrict__ qb,
    __bf16* __restrict__ kb, __bf16* __restrict__ vb) {
  __shared__ alignas(16) __bf16 ldsA[128 * kLD];
  __shared__ alignas(16) __bf16 ldsBT[128 * kLD];
  v8f acc[2][4];
  const int bm = blockIdx.y * 128;
  const int bn = blockIdx.x * 128;
  gemm_core_128x128(xb, wab, 3 * kC, kC, bm, bn, ldsA, ldsBT, acc);

  const int tid  = threadIdx.x;
  const int lane = tid & 31;
  const int wave = tid >> 5;
  const int wm = wave >> 1, wn = wave & 1;
  const int kh = lane >> 4, lm = lane & 15;
#pragma unroll
  for (int im = 0; im < 2; ++im) {
#pragma unroll
    for (int in = 0; in < 4; ++in) {
      const int   gn   = bn + wn * 64 + in * 16 + lm;
      const float bias = b_attn[gn];
      const int   sec  = gn >> 9;  // 0=q 1=k 2=v
      const int   c    = gn & (kC - 1);
      const int   h    = c >> 5;
      const int   d    = c & (kD - 1);
      __bf16*     dst  = (sec == 0) ? qb : ((sec == 1) ? kb : vb);
      const float scl  = (sec == 0) ? QSCALE : 1.0f;
#pragma unroll
      for (int i = 0; i < 8; ++i) {
        const int   gm  = bm + wm * 32 + im * 16 + i + 8 * kh;
        const int   b   = gm >> 11;
        const int   t   = gm & (kT - 1);
        const float val = (acc[im][in][i] + bias) * scl;
        dst[((size_t)(b * kH + h) * kT + t) * kD + d] = (__bf16)val;
      }
    }
  }
}

// ---------------------------------------------------------------------------
// Flash attention, one (b,h,128-query block) per workgroup; each wave owns a
// 16-query stripe. D=32 == bf16 WMMA K, so S=Q*K^T is one WMMA per 16x16 tile.
// K tile staged via async DMA; V transposed manually; P staged per-wave.
// ---------------------------------------------------------------------------
__global__ void __launch_bounds__(256) attn_kernel(const __bf16* __restrict__ Qg,
                                                   const __bf16* __restrict__ Kg,
                                                   const __bf16* __restrict__ Vg,
                                                   __bf16* __restrict__ Yg) {
  constexpr int LDK = 48;  // 96B rows
  constexpr int LDV = 72;  // 144B rows
  constexpr int LDP = 72;
  __shared__ alignas(16) __bf16 ldsK[64 * LDK];     // K tile [64 keys][32]
  __shared__ alignas(16) __bf16 ldsVt[32 * LDV];    // V^T tile [32 dims][64 keys]
  __shared__ alignas(16) __bf16 ldsP[8][16 * LDP];  // per-wave P [16 q][64 keys]

  const int h = blockIdx.y, b = blockIdx.z;
  const int tid = threadIdx.x, lane = tid & 31, wave = tid >> 5;
  const int kh = lane >> 4, lm = lane & 15;
  const size_t  head = ((size_t)(b * kH + h)) * kT * kD;
  const __bf16* Q = Qg + head;
  const __bf16* K = Kg + head;
  const __bf16* V = Vg + head;
  const int q0 = blockIdx.x * 128 + wave * 16;

  const v16bf qf = load_a_frag(Q + (size_t)(q0 + lm) * kD, kh);

  float mrun[8], lrun[8];
  v8f   o0 = vzero8(), o1 = vzero8();
#pragma unroll
  for (int i = 0; i < 8; ++i) {
    mrun[i] = -1e30f;
    lrun[i] = 0.0f;
  }

  const int      r    = tid >> 2;            // 0..63 tile row per thread
  const int      e0   = (tid & 3) * 8;       // element offset within row
  const unsigned klds = lds_off_of(ldsK + r * LDK + e0);

  for (int kt = 0; kt < kT; kt += 64) {
    {  // K tile via async DMA: one 16B chunk per thread
      const unsigned goff =
          (unsigned)(((size_t)(kt + r) * kD + e0) * sizeof(__bf16));
      async_load_b128(klds, goff, K);
    }
    {  // V tile, transposed into [dim][key]
      const __bf16* g = V + (size_t)(kt + r) * kD + e0;
#pragma unroll
      for (int j = 0; j < 8; ++j) ldsVt[(e0 + j) * LDV + r] = g[j];
    }
    if (kt + 64 < kT) {
      __builtin_prefetch(K + (size_t)(kt + 64 + r) * kD + e0, 0, 0);
      __builtin_prefetch(V + (size_t)(kt + 64 + r) * kD + e0, 0, 0);
    }
    wait_asynccnt0();
    __syncthreads();

    // S = (q*scale) . K^T  for 4 key subtiles of 16
    v8f s[4];
#pragma unroll
    for (int jt = 0; jt < 4; ++jt) {
      const v16bf kf = load_b_frag(ldsK + (jt * 16 + lm) * LDK, kh);
      s[jt] = wmma_bf16(qf, kf, vzero8());
    }

    // row-wise max across the 64 keys (16 lanes per half-wave share a row set)
    float mx[8];
#pragma unroll
    for (int i = 0; i < 8; ++i)
      mx[i] = fmaxf(fmaxf(s[0][i], s[1][i]), fmaxf(s[2][i], s[3][i]));
#pragma unroll
    for (int off = 1; off < 16; off <<= 1)
#pragma unroll
      for (int i = 0; i < 8; ++i) mx[i] = fmaxf(mx[i], __shfl_xor(mx[i], off, 32));

    float rs[8];
#pragma unroll
    for (int i = 0; i < 8; ++i) {
      const float mn    = fmaxf(mrun[i], mx[i]);
      const float alpha = exp2f(mrun[i] - mn);
      mrun[i] = mn;
      o0[i] *= alpha;
      o1[i] *= alpha;
      lrun[i] *= alpha;
      rs[i] = 0.0f;
    }

    // P = exp2(S - m), staged through this wave's private LDS tile (same-wave
    // DS ops are in-order -> no barrier needed before the A-frag reloads)
    __bf16* P = &ldsP[wave][0];
#pragma unroll
    for (int jt = 0; jt < 4; ++jt)
#pragma unroll
      for (int i = 0; i < 8; ++i) {
        const float pv = exp2f(s[jt][i] - mrun[i]);
        rs[i] += pv;
        P[(i + 8 * kh) * LDP + jt * 16 + lm] = (__bf16)pv;
      }
#pragma unroll
    for (int off = 1; off < 16; off <<= 1)
#pragma unroll
      for (int i = 0; i < 8; ++i) rs[i] += __shfl_xor(rs[i], off, 32);
#pragma unroll
    for (int i = 0; i < 8; ++i) lrun[i] += rs[i];

    // O += P . V  (two K-chunks of 32 keys, two 16-dim output tiles)
#pragma unroll
    for (int kc = 0; kc < 2; ++kc) {
      const v16bf pf  = load_a_frag(P + lm * LDP + kc * 32, kh);
      const v16bf vf0 = load_b_frag(ldsVt + lm * LDV + kc * 32, kh);
      const v16bf vf1 = load_b_frag(ldsVt + (16 + lm) * LDV + kc * 32, kh);
      o0 = wmma_bf16(pf, vf0, o0);
      o1 = wmma_bf16(pf, vf1, o1);
    }
    __syncthreads();
  }

#pragma unroll
  for (int i = 0; i < 8; ++i) {
    const float inv = 1.0f / lrun[i];
    o0[i] *= inv;
    o1[i] *= inv;
  }
  // merge heads: y[b][t][h*32 + d] (bf16) for the final projection
  __bf16* yrow = Yg + ((size_t)b * kT) * kC + h * kD;
#pragma unroll
  for (int i = 0; i < 8; ++i) {
    const int t = q0 + i + 8 * kh;
    yrow[(size_t)t * kC + lm]      = (__bf16)o0[i];
    yrow[(size_t)t * kC + 16 + lm] = (__bf16)o1[i];
  }
}

// ---------------------------------------------------------------------------
// Output projection: [8192,512] x [512,512] + bias -> fp32 out
// ---------------------------------------------------------------------------
__global__ void __launch_bounds__(256) proj_gemm_kernel(
    const __bf16* __restrict__ yb, const __bf16* __restrict__ wpb,
    const float* __restrict__ b_proj, float* __restrict__ out) {
  __shared__ alignas(16) __bf16 ldsA[128 * kLD];
  __shared__ alignas(16) __bf16 ldsBT[128 * kLD];
  v8f acc[2][4];
  const int bm = blockIdx.y * 128;
  const int bn = blockIdx.x * 128;
  gemm_core_128x128(yb, wpb, kC, kC, bm, bn, ldsA, ldsBT, acc);

  const int tid  = threadIdx.x;
  const int lane = tid & 31;
  const int wave = tid >> 5;
  const int wm = wave >> 1, wn = wave & 1;
  const int kh = lane >> 4, lm = lane & 15;
#pragma unroll
  for (int im = 0; im < 2; ++im) {
#pragma unroll
    for (int in = 0; in < 4; ++in) {
      const int   gn   = bn + wn * 64 + in * 16 + lm;
      const float bias = b_proj[gn];
#pragma unroll
      for (int i = 0; i < 8; ++i) {
        const int gm = bm + wm * 32 + im * 16 + i + 8 * kh;
        out[(size_t)gm * kC + gn] = acc[im][in][i] + bias;
      }
    }
  }
}

extern "C" void kernel_launch(void* const* d_in, const int* in_sizes, int n_in,
                              void* d_out, int out_size, void* d_ws, size_t ws_size,
                              hipStream_t stream) {
  (void)in_sizes; (void)n_in; (void)out_size; (void)ws_size;
  const float* x      = (const float*)d_in[0];
  const float* w_attn = (const float*)d_in[1];
  const float* b_attn = (const float*)d_in[2];
  const float* w_proj = (const float*)d_in[3];
  const float* b_proj = (const float*)d_in[4];
  float*       out    = (float*)d_out;

  // workspace carve-up (bf16), ~44 MB total
  __bf16* p   = (__bf16*)d_ws;
  __bf16* xb  = p; p += (size_t)kRows * kC;
  __bf16* wab = p; p += (size_t)kC * 3 * kC;
  __bf16* wpb = p; p += (size_t)kC * kC;
  __bf16* qb  = p; p += (size_t)kRows * kC;
  __bf16* kb  = p; p += (size_t)kRows * kC;
  __bf16* vb  = p; p += (size_t)kRows * kC;
  __bf16* yb  = p; p += (size_t)kRows * kC;

  {
    int n = kRows * kC;
    cast_f32_to_bf16<<<(n + 255) / 256, 256, 0, stream>>>(x, xb, n);
  }
  {
    int n = kC * 3 * kC;
    cast_f32_to_bf16<<<(n + 255) / 256, 256, 0, stream>>>(w_attn, wab, n);
  }
  {
    int n = kC * kC;
    cast_f32_to_bf16<<<(n + 255) / 256, 256, 0, stream>>>(w_proj, wpb, n);
  }

  qkv_gemm_kernel<<<dim3(3 * kC / 128, kRows / 128), 256, 0, stream>>>(
      xb, wab, b_attn, qb, kb, vb);
  attn_kernel<<<dim3(kT / 128, kH, kB), 256, 0, stream>>>(qb, kb, vb, yb);
  proj_gemm_kernel<<<dim3(kC / 128, kRows / 128), 256, 0, stream>>>(
      yb, wpb, b_proj, out);
}
